// FractalResonanceTransformer_31877247270958
// MI455X (gfx1250) — matrix operand order repaired and checked by reading, not compiled, not run on hardware
//
#include <hip/hip_runtime.h>
#include <hip/hip_bf16.h>
#include <math.h>

// ---------------------------------------------------------------------------
// FractalResonanceTransformer for MI455X (gfx1250, wave32)
//   Phase 1: input projections as bf16 WMMA GEMMs (v_wmma_f32_16x16x32_bf16)
//   Phase 2: latency-hiding sequential scan, one lane per (b,d,n) state.
//            Inner loop minimized: |h_c| and |inj| are known analytically, so
//            the resonance gate needs no extra sqrts; fast rcp-based divides.
// ---------------------------------------------------------------------------

typedef __attribute__((ext_vector_type(16))) __bf16 v16bf;
typedef __attribute__((ext_vector_type(8)))  float  v8f;

#define KAPPA 0.5f

// C(MxN) = act( A(MxK,f32) * B(KxN,f32) ), bf16 WMMA, one 16x16 tile per wave.
// M % 16 == 0, N % 16 == 0, K % 32 == 0.   act: 0 = identity, 1 = softplus
__global__ __launch_bounds__(256) void wmma_gemm_bf16(
    const float* __restrict__ A, const float* __restrict__ B,
    float* __restrict__ C, int M, int N, int K, int act)
{
    const int wave = (blockIdx.x * blockDim.x + threadIdx.x) >> 5;
    const int lane = threadIdx.x & 31;
    const int ntile = N >> 4;
    const int tiles = (M >> 4) * ntile;
    if (wave >= tiles) return;                 // wave-uniform: EXEC all-1s for WMMA
    const int ti = wave / ntile;
    const int tj = wave - ti * ntile;
    const int m  = lane & 15;
    const bool hih = (lane >= 16);

    const float* __restrict__ Arow = A + (size_t)(ti * 16 + m) * K;
    const float* __restrict__ Bcol = B + (tj * 16 + m);

    v8f acc = {};
    for (int k0 = 0; k0 < K; k0 += 32) {
        v16bf a, b;
#pragma unroll
        for (int e = 0; e < 16; ++e) {
            // ISA 7.12.2: 16-bit A 16x32 per-lane K pattern
            const int ka = (e < 8 ? e : e + 8) + (hih ? 8 : 0);
            // B 32x16: same per-lane K pattern as A
            const int kb = (e < 8 ? e : e + 8) + (hih ? 8 : 0);
            a[e] = (__bf16)Arow[k0 + ka];
            b[e] = (__bf16)Bcol[(size_t)(k0 + kb) * N];
        }
        acc = __builtin_amdgcn_wmma_f32_16x16x32_bf16(
            false, a, false, b, (short)0, acc, false, false);
    }

#pragma unroll
    for (int v = 0; v < 8; ++v) {
        float val = acc[v];
        if (act == 1)  // softplus, fast-path transcendentals
            val = (val > 20.0f) ? val : __logf(1.0f + __expf(val));
        const int row = ti * 16 + v + (hih ? 8 : 0);
        C[(size_t)row * N + tj * 16 + m] = val;
    }
}

// Sequential recurrence. One lane per (b, d, n): lane holds one complex state.
// Wave = 32 lanes = 2 adjacent d-channels x 16 n-states. n-reduction via
// 16-lane xor shuffles. 32768 threads total = 1024 wave32s (latency hiding).
//
// Per-step math (algebraically reduced from the reference):
//   h_c   = (cm/|h|) * h          with cm = min(|h|^e, 10)     [phase preserved]
//   cosΔφ = u*(cosφ·hcr + sinφ·hci) / (|u|·cm)                 [|h_c|=cm, |inj|=|u|]
//   gamma = (KAPPA/2)*(1+cosΔφ)
//   h'    = (tanh(|s|)/|s|) * s   with s = h_c + gamma*inj     [phase preserved]
__global__ __launch_bounds__(256) void frt_scan(
    const float* __restrict__ x,      // (B,L,D)
    const float* __restrict__ A_log,  // (N)
    const float* __restrict__ delta,  // (B,L,D)
    const float* __restrict__ Bt,     // (B,L,N)
    const float* __restrict__ Ct,     // (B,L,N)
    const float* __restrict__ D_skip, // (D)
    float* __restrict__ out)          // (B,L,D)
{
    const int L = 1024, D = 1024, N = 16;
    const int gt = blockIdx.x * blockDim.x + threadIdx.x;
    const int n  = gt & 15;
    const int bd = gt >> 4;
    const int d  = bd & (D - 1);
    const int b  = bd >> 10;

    const float A_n  = __expf(A_log[n]);
    const float phi  = 6.283185307179586f * (float)n / (float)N;
    float sphi, cphi;
    __sincosf(phi, &sphi, &cphi);
    const float Dsk = D_skip[d];

    // dna_init: magnitude 0.01 on carrier phase
    float hr = 0.01f * cphi;
    float hi = 0.01f * sphi;

    const size_t baseBD = (size_t)b * L * D + d;
    const size_t baseBN = (size_t)b * L * N + n;

    for (int t = 0; t < L; ++t) {
        const float x_t = x[baseBD + (size_t)t * D];
        const float d_t = delta[baseBD + (size_t)t * D];
        const float Bn  = Bt[baseBN + (size_t)t * N];
        const float Cn  = Ct[baseBN + (size_t)t * N];

        // fractal_compress: |h| -> min(|h|^e, 10), phase preserved
        const float e    = 1.0f + d_t * A_n;
        const float mag  = __builtin_sqrtf(hr * hr + hi * hi + 1e-8f);
        const float cm   = fminf(__expf(e * __logf(mag + 1e-8f)), 10.0f);
        const float sc   = __fdividef(cm, mag);
        const float hcr  = sc * hr;
        const float hci  = sc * hi;

        // resonance gate: cos^2(dphi/2) = (1 + cosd)/2;
        // |h_c| = cm and |inj| = |u| analytically -> no extra sqrts.
        const float u    = x_t * Bn;
        const float dotn = cphi * hcr + sphi * hci;   // = cm * cos(phi_hc - phi_n)
        const float cosd = __fdividef(u * dotn, fabsf(u) * cm + 1e-20f);
        const float gamma = (0.5f * KAPPA) * (1.0f + cosd);

        // accumulate + mag_squash: |s| -> tanh(|s|), phase preserved
        const float gu = gamma * u;
        const float sr = hcr + gu * cphi;
        const float si = hci + gu * sphi;
        const float ms = __builtin_sqrtf(sr * sr + si * si + 1e-8f);
        const float ts = __fdividef(tanhf(ms), ms);
        hr = ts * sr;
        hi = ts * si;

        // y_t[b,d] = sum_n hr[n] * C_t[b,n]  (16-lane xor-tree reduction)
        float y = hr * Cn;
        y += __shfl_xor(y, 8, 32);
        y += __shfl_xor(y, 4, 32);
        y += __shfl_xor(y, 2, 32);
        y += __shfl_xor(y, 1, 32);
        if (n == 0)
            out[baseBD + (size_t)t * D] = y + x_t * Dsk;
    }
}

extern "C" void kernel_launch(void* const* d_in, const int* in_sizes, int n_in,
                              void* d_out, int out_size, void* d_ws, size_t ws_size,
                              hipStream_t stream) {
    const float* x      = (const float*)d_in[0];  // (2,1024,1024)
    const float* A_log  = (const float*)d_in[1];  // (16)
    const float* W_dt1  = (const float*)d_in[2];  // (1024,64)
    const float* W_dt2  = (const float*)d_in[3];  // (64,1024)
    const float* W_B    = (const float*)d_in[4];  // (1024,16)
    const float* W_C    = (const float*)d_in[5];  // (1024,16)
    const float* D_skip = (const float*)d_in[6];  // (1024)
    float* out = (float*)d_out;

    const int BL = 2048, Dm = 1024, R = 64, N = 16;

    char*  ws    = (char*)d_ws;
    float* T1    = (float*)ws;                                      // 2048x64
    float* delta = (float*)(ws + (size_t)BL * R * 4);               // 2048x1024
    float* Btp   = (float*)(ws + (size_t)BL * R * 4 + (size_t)BL * Dm * 4); // 2048x16
    float* Ctp   = Btp + (size_t)BL * N;                            // 2048x16

    auto gemm = [&](const float* A, const float* B, float* C,
                    int M, int Nn, int K, int act) {
        const int tiles   = (M / 16) * (Nn / 16);
        const int threads = tiles * 32;
        wmma_gemm_bf16<<<(threads + 255) / 256, 256, 0, stream>>>(A, B, C, M, Nn, K, act);
    };

    gemm(x,  W_dt1, T1,    BL, R,  Dm, 0);  // T1 = x @ W_dt1
    gemm(T1, W_dt2, delta, BL, Dm, R,  1);  // delta = softplus(T1 @ W_dt2)
    gemm(x,  W_B,   Btp,   BL, N,  Dm, 0);  // Bt
    gemm(x,  W_C,   Ctp,   BL, N,  Dm, 0);  // Ct

    frt_scan<<<128, 256, 0, stream>>>(x, A_log, delta, Btp, Ctp, D_skip, out);
}
